// Net_4733053960821
// MI455X (gfx1250) — matrix operand order repaired and checked by reading, not compiled, not run on hardware
//
#include <hip/hip_runtime.h>
#include <hip/hip_bf16.h>

// ---------------------------------------------------------------------------
// Problem constants (from reference)
// ---------------------------------------------------------------------------
#define NN 4096      // nodes
#define EE 8192      // edges
#define GG 256       // graphs
#define DD 128       // DIM

typedef __attribute__((ext_vector_type(16))) _Float16 v16h;
typedef __attribute__((ext_vector_type(8)))  _Float16 v8h;
typedef __attribute__((ext_vector_type(4)))  _Float16 v4h;
typedef __attribute__((ext_vector_type(8)))  float    v8f;
typedef __attribute__((ext_vector_type(4)))  int      v4i;

__device__ __forceinline__ float sigmf(float x) { return 1.0f / (1.0f + expf(-x)); }

// ---------------------------------------------------------------------------
// CDNA5 async global->LDS copy (ASYNCcnt path), with graceful fallback.
// Probe result (round 2): builtin exists and takes AS(1)/AS(3) pointers to
// 16-byte int vectors.
// ---------------------------------------------------------------------------
#if __has_builtin(__builtin_amdgcn_global_load_async_to_lds_b128)
#define HAVE_ASYNC_LDS 1
#else
#define HAVE_ASYNC_LDS 0
#endif

__device__ __forceinline__ void async_cp_b128(const _Float16* g, _Float16* l) {
#if HAVE_ASYNC_LDS
    __builtin_amdgcn_global_load_async_to_lds_b128(
        (__attribute__((address_space(1))) v4i*)g,
        (__attribute__((address_space(3))) v4i*)l, 0, 0);
#else
    *(v8h*)l = *(const v8h*)g;
#endif
}

__device__ __forceinline__ void async_wait0() {
#if HAVE_ASYNC_LDS
#if __has_builtin(__builtin_amdgcn_s_wait_asynccnt)
    __builtin_amdgcn_s_wait_asynccnt(0);
#else
    asm volatile("s_wait_asynccnt 0" ::: "memory");
#endif
#endif
}

// ---------------------------------------------------------------------------
// Elementwise helpers
// ---------------------------------------------------------------------------
__global__ void k_zero_f32(float* p, int n) {
    int i = blockIdx.x * 256 + threadIdx.x;
    if (i < n) p[i] = 0.0f;
}

// f32 -> f16 with zero padding: src is [Ms,Ks] row-major, dst is [Md,Kd]
__global__ void k_cvt_f16(const float* __restrict__ s, _Float16* __restrict__ d,
                          int Md, int Kd, int Ms, int Ks) {
    int i = blockIdx.x * 256 + threadIdx.x;
    if (i >= Md * Kd) return;
    int m = i / Kd, k = i - m * Kd;
    float v = (m < Ms && k < Ks) ? s[(size_t)m * Ks + k] : 0.0f;
    d[i] = (_Float16)v;
}

// ---------------------------------------------------------------------------
// Generic WMMA GEMM:  out = epilogue( A[M,K] @ W[N,K]^T )
// A, W row-major f16; K % 32 == 0; N % (16*NT) == 0; M % 128 == 0.
// Block: 256 thr = 8 waves stacked in M (128 rows) x (16*NT) cols.
// B tile (16*NT x 32) is staged in LDS via async copies, double-buffered:
// the async engine fills tile k+1 while WMMAs consume tile k from LDS.
// LDS row stride padded to 40 halves -> conflict-free ds_load_b128 frags.
// Epilogue: optional inference-BN (g,b,m,v), optional bias, optional ReLU,
//           writes f32 and/or f16.
// ---------------------------------------------------------------------------
__device__ __forceinline__ v16h loadA_frag(const _Float16* __restrict__ p, int half8) {
    // A 16x32 f16 layout: lanes 0-15 K={0..7,16..23}; lanes 16-31 K={8..15,24..31}
    v8h lo = *(const v8h*)(p + half8);
    v8h hi = *(const v8h*)(p + 16 + half8);
    v16h r;
#pragma unroll
    for (int j = 0; j < 8; ++j) { r[j] = lo[j]; r[8 + j] = hi[j]; }
    return r;
}

template <int NT>
__global__ void k_gemm_wmma(const _Float16* __restrict__ A,
                            const _Float16* __restrict__ W,
                            float* __restrict__ out32,
                            _Float16* __restrict__ out16,
                            const float* __restrict__ bn_g,
                            const float* __restrict__ bn_b,
                            const float* __restrict__ bn_m,
                            const float* __restrict__ bn_v,
                            const float* __restrict__ bias,
                            int M, int N, int K, int relu) {
    constexpr int TN   = 16 * NT;   // tile columns
    constexpr int BROW = 40;        // padded LDS row stride in halves (80 B)
    __shared__ _Float16 Bs[2][TN * BROW];

    const int tid  = threadIdx.x;
    const int lane = tid & 31;
    const int wave = tid >> 5;                       // 0..7
    const int n0   = blockIdx.x * TN;
    const int m0   = (blockIdx.y * 8 + wave) * 16;

    const int half8 = (lane >> 4) * 8;               // A fragment K sub-offset
    const int q16   = (lane >> 4) * 8;               // B fragment K sub-offset

    // cooperative B-tile copy role: thread -> (row, 8-half chunk)
    const int  crow = tid >> 2;                      // 0 .. TN-1 (for NT=4)
    const int  cchk = (tid & 3) * 8;                 // half offset within row
    const bool docp = (tid < TN * 4);
    const _Float16* gw  = W + (size_t)(n0 + crow) * K + cchk;
    _Float16* lw0 = &Bs[0][crow * BROW + cchk];
    _Float16* lw1 = &Bs[1][crow * BROW + cchk];

    // prologue: stage tile 0
    if (docp) async_cp_b128(gw, lw0);

    v8f acc[NT];
    v8f zero = {};
#pragma unroll
    for (int t = 0; t < NT; ++t) acc[t] = zero;

    const _Float16* pa = A + (size_t)(m0 + (lane & 15)) * K;
    const int nk = K >> 5;
    for (int s = 0; s < nk; ++s) {
        async_wait0();
        __syncthreads();                             // tile s visible in LDS
        if (s + 1 < nk && docp)
            async_cp_b128(gw + (s + 1) * 32, ((s + 1) & 1) ? lw1 : lw0);
        v16h a = loadA_frag(pa + s * 32, half8);
        const _Float16* bbase = &Bs[s & 1][(lane & 15) * BROW + q16];
#pragma unroll
        for (int t = 0; t < NT; ++t) {
            v8h blo = *(const v8h*)(bbase + t * 16 * BROW);
            v8h bhi = *(const v8h*)(bbase + t * 16 * BROW + 16);
            v16h b;
#pragma unroll
            for (int j = 0; j < 8; ++j) { b[j] = blo[j]; b[8 + j] = bhi[j]; }
            acc[t] = __builtin_amdgcn_wmma_f32_16x16x32_f16(
                false, a, false, b, (short)0, acc[t], false, false);
        }
        __syncthreads();                             // done reading tile s
    }

    // Epilogue. C/D layout: lane -> col = lane&15 ; VGPR v -> row = v + 8*(lane>>4)
    const int col_lo    = lane & 15;
    const int mrow_base = m0 + ((lane >> 4) * 8);
#pragma unroll
    for (int t = 0; t < NT; ++t) {
        int n = n0 + t * 16 + col_lo;
        float sc = 1.0f, sh = 0.0f;
        if (bn_g) {
            float inv = rsqrtf(bn_v[n] + 1e-5f);
            sc = bn_g[n] * inv;
            sh = bn_b[n] - bn_m[n] * sc;
        }
        if (bias) sh += bias[n];
#pragma unroll
        for (int v = 0; v < 8; ++v) {
            int mrow = mrow_base + v;
            float val = acc[t][v] * sc + sh;
            if (relu) val = fmaxf(val, 0.0f);
            size_t idx = (size_t)mrow * N + n;
            if (out32) out32[idx] = val;
            if (out16) out16[idx] = (_Float16)val;
        }
    }
}

// ---------------------------------------------------------------------------
// Degree (scatter count over dst)
// ---------------------------------------------------------------------------
__global__ void k_deg(const int* __restrict__ dst, float* __restrict__ deg, int E) {
    int e = blockIdx.x * 256 + threadIdx.x;
    if (e < E) atomicAdd(&deg[dst[e]], 1.0f);
}

// ---------------------------------------------------------------------------
// NNConv message + scatter-sum. One wave32 per edge; lane owns 4 output feats.
// msg[f] = sum_d out[src,d] * w[e,d,f] ; atomically accumulated into agg[dst].
// w stored f16, one 128-f16 row (256 B) streamed per inner step -> HBM bound.
// ---------------------------------------------------------------------------
__global__ void k_msg_scatter(const float* __restrict__ out,
                              const _Float16* __restrict__ w,
                              const int* __restrict__ src,
                              const int* __restrict__ dst,
                              float* __restrict__ agg, int E) {
    int e    = blockIdx.x * 8 + (threadIdx.x >> 5);
    int lane = threadIdx.x & 31;
    if (e >= E) return;
    int s = src[e], d = dst[e];
    const float*    op = out + (size_t)s * DD;
    const _Float16* wp = w + (size_t)e * (DD * DD) + lane * 4;
    float a0 = 0.f, a1 = 0.f, a2 = 0.f, a3 = 0.f;
    for (int dd = 0; dd < DD; ++dd) {
        if ((dd & 15) == 0 && dd + 16 < DD)
            __builtin_prefetch(wp + (size_t)(dd + 16) * DD, 0, 1);
        float o  = op[dd];
        v4h   wv = *(const v4h*)(wp + (size_t)dd * DD);
        a0 += o * (float)wv[0];
        a1 += o * (float)wv[1];
        a2 += o * (float)wv[2];
        a3 += o * (float)wv[3];
    }
    float* ap = agg + (size_t)d * DD + lane * 4;
    atomicAdd(ap + 0, a0);
    atomicAdd(ap + 1, a1);
    atomicAdd(ap + 2, a2);
    atomicAdd(ap + 3, a3);
}

// agg / max(deg,1) + conv_bias, ReLU, store as f16 (feeds GRU gx GEMM)
__global__ void k_agg_finalize(const float* __restrict__ agg,
                               const float* __restrict__ deg,
                               const float* __restrict__ cbias,
                               _Float16* __restrict__ m16, int n) {
    int i = blockIdx.x * 256 + threadIdx.x;
    if (i >= n) return;
    int node = i >> 7, d = i & (DD - 1);
    float v = agg[i] / fmaxf(deg[node], 1.0f) + cbias[d];
    m16[i] = (_Float16)fmaxf(v, 0.0f);
}

// GRU cell pointwise combine; h updated in f32 and mirrored in f16
__global__ void k_gru_combine(const float* __restrict__ gx,
                              const float* __restrict__ gh,
                              float* __restrict__ h,
                              _Float16* __restrict__ h16, int n) {
    int i = blockIdx.x * 256 + threadIdx.x;
    if (i >= n) return;
    int node = i >> 7, d = i & (DD - 1);
    const float* px = gx + (size_t)node * (3 * DD);
    const float* ph = gh + (size_t)node * (3 * DD);
    float r  = sigmf(px[d] + ph[d]);
    float z  = sigmf(px[DD + d] + ph[DD + d]);
    float nn = tanhf(px[2 * DD + d] + r * ph[2 * DD + d]);
    float hv = (1.0f - z) * nn + z * h[i];
    h[i]   = hv;
    h16[i] = (_Float16)hv;
}

// ---------------------------------------------------------------------------
// Graph CSR boundaries from sorted batch vector
// ---------------------------------------------------------------------------
__global__ void k_bounds_init(int* gstart, int* gend, int G) {
    int g = blockIdx.x * 256 + threadIdx.x;
    if (g < G) { gstart[g] = 0; gend[g] = 0; }
}
__global__ void k_bounds_find(const int* __restrict__ batch, int* gstart, int* gend, int N) {
    int i = blockIdx.x * 256 + threadIdx.x;
    if (i >= N) return;
    int b = batch[i];
    if (i == 0 || batch[i - 1] != b) gstart[b] = i;
    if (i == N - 1 || batch[i + 1] != b) gend[b] = i + 1;
}

// ---------------------------------------------------------------------------
// Set2Set LSTM: gates then pointwise update (G=256 is tiny -> VALU)
// ---------------------------------------------------------------------------
__global__ void k_lstm_gates(const float* __restrict__ qstar,
                             const float* __restrict__ hq,
                             const float* __restrict__ Wih,
                             const float* __restrict__ Whh,
                             const float* __restrict__ bih,
                             const float* __restrict__ bhh,
                             float* __restrict__ gates) {
    int idx = blockIdx.x * 256 + threadIdx.x;     // G * 4*DD
    if (idx >= GG * 4 * DD) return;
    int g = idx >> 9, j = idx & (4 * DD - 1);
    const float* q  = qstar + (size_t)g * (2 * DD);
    const float* wq = Wih + (size_t)j * (2 * DD);
    float acc = bih[j] + bhh[j];
    for (int k = 0; k < 2 * DD; ++k) acc += q[k] * wq[k];
    const float* hp = hq + (size_t)g * DD;
    const float* wh = Whh + (size_t)j * DD;
    for (int k = 0; k < DD; ++k) acc += hp[k] * wh[k];
    gates[idx] = acc;
}

__global__ void k_lstm_update(const float* __restrict__ gates,
                              float* __restrict__ c,
                              float* __restrict__ hq, int n) {
    int i = blockIdx.x * 256 + threadIdx.x;       // G * DD
    if (i >= n) return;
    int g = i >> 7, d = i & (DD - 1);
    const float* gp = gates + (size_t)g * (4 * DD);
    float ig = sigmf(gp[d]);
    float fg = sigmf(gp[DD + d]);
    float gg = tanhf(gp[2 * DD + d]);
    float og = sigmf(gp[3 * DD + d]);
    float cv = fg * c[i] + ig * gg;
    c[i]  = cv;
    hq[i] = og * tanhf(cv);
}

// e[i] = dot(out[i], hq[batch[i]]) ; one wave per node, shfl_xor reduce (wave32)
__global__ void k_attn_e(const float* __restrict__ out,
                         const float* __restrict__ hq,
                         const int* __restrict__ batch,
                         float* __restrict__ e, int N) {
    int i    = blockIdx.x * 8 + (threadIdx.x >> 5);
    int lane = threadIdx.x & 31;
    if (i >= N) return;
    const float* op = out + (size_t)i * DD + lane * 4;
    const float* qp = hq + (size_t)batch[i] * DD + lane * 4;
    float a = op[0] * qp[0] + op[1] * qp[1] + op[2] * qp[2] + op[3] * qp[3];
    for (int off = 16; off > 0; off >>= 1) a += __shfl_xor(a, off, 32);
    if (lane == 0) e[i] = a;
}

// Per-graph softmax + weighted sum, writes q_star = [hq, r]. Block 128 / graph.
__global__ void k_set2set_reduce(const float* __restrict__ e,
                                 const float* __restrict__ out,
                                 const float* __restrict__ hq,
                                 const int* __restrict__ gstart,
                                 const int* __restrict__ gend,
                                 float* __restrict__ qstar) {
    int g = blockIdx.x, tid = threadIdx.x;
    __shared__ float red[DD];
    __shared__ float s_mx, s_inv;
    int s = gstart[g], en = gend[g];
    // max
    float mx = -3.4e38f;
    for (int j = s + tid; j < en; j += DD) mx = fmaxf(mx, e[j]);
    red[tid] = mx; __syncthreads();
    for (int st = DD / 2; st > 0; st >>= 1) {
        if (tid < st) red[tid] = fmaxf(red[tid], red[tid + st]);
        __syncthreads();
    }
    if (tid == 0) s_mx = red[0];
    __syncthreads();
    mx = s_mx;
    // sum of exp
    float sm = 0.0f;
    for (int j = s + tid; j < en; j += DD) sm += expf(e[j] - mx);
    red[tid] = sm; __syncthreads();
    for (int st = DD / 2; st > 0; st >>= 1) {
        if (tid < st) red[tid] += red[tid + st];
        __syncthreads();
    }
    if (tid == 0) s_inv = (en > s) ? (1.0f / red[0]) : 0.0f;
    __syncthreads();
    float inv = s_inv;
    // r_d : thread tid owns feature d = tid
    float r = 0.0f;
    for (int j = s; j < en; ++j)
        r += expf(e[j] - mx) * inv * out[(size_t)j * DD + tid];
    qstar[(size_t)g * (2 * DD) + tid]      = hq[(size_t)g * DD + tid];
    qstar[(size_t)g * (2 * DD) + DD + tid] = r;
}

// feat = concat(out[t0], out[t1], q_star[batch]) -> f16 [N, 4*DD]
__global__ void k_feat_gather(const float* __restrict__ h,
                              const float* __restrict__ qstar,
                              const int* __restrict__ t0,
                              const int* __restrict__ t1,
                              const int* __restrict__ batch,
                              _Float16* __restrict__ feat, int n) {
    int idx = blockIdx.x * 256 + threadIdx.x;     // N * 4*DD
    if (idx >= n) return;
    int i = idx >> 9, d = idx & (4 * DD - 1);
    float v;
    if (d < DD)            v = h[(size_t)t0[i] * DD + d];
    else if (d < 2 * DD)   v = h[(size_t)t1[i] * DD + (d - DD)];
    else                   v = qstar[(size_t)batch[i] * (2 * DD) + (d - 2 * DD)];
    feat[idx] = (_Float16)v;
}

// out[i] = p3[i, class[i]] + b3[class[i]]   (p3 padded to 16 cols)
__global__ void k_final_gather(const float* __restrict__ p3,
                               const float* __restrict__ b3,
                               const int* __restrict__ cls,
                               float* __restrict__ out, int N) {
    int i = blockIdx.x * 256 + threadIdx.x;
    if (i >= N) return;
    int c = cls[i];
    out[i] = p3[(size_t)i * 16 + c] + b3[c];
}

// ---------------------------------------------------------------------------
// Host side
// ---------------------------------------------------------------------------
static inline dim3 g1(int n) { return dim3((n + 255) / 256); }

static void launch_gemm(hipStream_t st, const _Float16* A, const _Float16* W,
                        float* o32, _Float16* o16,
                        const float* g, const float* b, const float* m, const float* v,
                        const float* bias, int M, int N, int K, int relu) {
    if (N % 64 == 0) {
        dim3 grid(N / 64, M / 128);
        k_gemm_wmma<4><<<grid, 256, 0, st>>>(A, W, o32, o16, g, b, m, v, bias, M, N, K, relu);
    } else {
        dim3 grid(N / 16, M / 128);
        k_gemm_wmma<1><<<grid, 256, 0, st>>>(A, W, o32, o16, g, b, m, v, bias, M, N, K, relu);
    }
}

extern "C" void kernel_launch(void* const* d_in, const int* in_sizes, int n_in,
                              void* d_out, int out_size, void* d_ws, size_t ws_size,
                              hipStream_t stream) {
    (void)in_sizes; (void)n_in; (void)out_size; (void)ws_size;

    // ---- inputs (setup_inputs dict order; params flattened depth-first) ----
    int ii = 0;
    const float* x          = (const float*)d_in[ii++]; // 0
    const int*   edge_index = (const int*)  d_in[ii++]; // 1  [2,E]
    const float* edge_attr  = (const float*)d_in[ii++]; // 2  [E,4]
    const int*   batch      = (const int*)  d_in[ii++]; // 3
    const int*   tindex     = (const int*)  d_in[ii++]; // 4  [2,N]
    const int*   tclass     = (const int*)  d_in[ii++]; // 5
    const float* pre_W1 = (const float*)d_in[ii++];
    const float *pre1g = (const float*)d_in[ii++], *pre1b = (const float*)d_in[ii++],
                *pre1m = (const float*)d_in[ii++], *pre1v = (const float*)d_in[ii++];
    const float* pre_W2 = (const float*)d_in[ii++];
    const float *pre2g = (const float*)d_in[ii++], *pre2b = (const float*)d_in[ii++],
                *pre2m = (const float*)d_in[ii++], *pre2v = (const float*)d_in[ii++];
    const float* enc_W1 = (const float*)d_in[ii++];
    const float *enc1g = (const float*)d_in[ii++], *enc1b = (const float*)d_in[ii++],
                *enc1m = (const float*)d_in[ii++], *enc1v = (const float*)d_in[ii++];
    const float* enc_W2 = (const float*)d_in[ii++];
    const float *enc2g = (const float*)d_in[ii++], *enc2b = (const float*)d_in[ii++],
                *enc2m = (const float*)d_in[ii++], *enc2v = (const float*)d_in[ii++];
    const float* enc_W3 = (const float*)d_in[ii++];
    const float *enc3g = (const float*)d_in[ii++], *enc3b = (const float*)d_in[ii++],
                *enc3m = (const float*)d_in[ii++], *enc3v = (const float*)d_in[ii++];
    const float* enc_W4 = (const float*)d_in[ii++];
    const float *enc4g = (const float*)d_in[ii++], *enc4b = (const float*)d_in[ii++],
                *enc4m = (const float*)d_in[ii++], *enc4v = (const float*)d_in[ii++];
    const float* conv_bias = (const float*)d_in[ii++];
    const float* gru_Wih = (const float*)d_in[ii++];
    const float* gru_Whh = (const float*)d_in[ii++];
    const float* gru_bih = (const float*)d_in[ii++];
    const float* gru_bhh = (const float*)d_in[ii++];
    const float* lstm_Wih = (const float*)d_in[ii++];
    const float* lstm_Whh = (const float*)d_in[ii++];
    const float* lstm_bih = (const float*)d_in[ii++];
    const float* lstm_bhh = (const float*)d_in[ii++];
    const float* pred_W1 = (const float*)d_in[ii++];
    const float *prd1g = (const float*)d_in[ii++], *prd1b = (const float*)d_in[ii++],
                *prd1m = (const float*)d_in[ii++], *prd1v = (const float*)d_in[ii++];
    const float* pred_W2 = (const float*)d_in[ii++];
    const float *prd2g = (const float*)d_in[ii++], *prd2b = (const float*)d_in[ii++],
                *prd2m = (const float*)d_in[ii++], *prd2v = (const float*)d_in[ii++];
    const float* pred_W3 = (const float*)d_in[ii++];
    const float* pred_b3 = (const float*)d_in[ii++];

    const int* e_src = edge_index;          // row 0
    const int* e_dst = edge_index + EE;     // row 1
    const int* t0 = tindex;
    const int* t1 = tindex + NN;

    // ---- workspace carve-up ----
    char*  base = (char*)d_ws;
    size_t off  = 0;
    auto alloc = [&](size_t bytes) -> void* {
        void* p = base + off;
        off += (bytes + 255) & ~(size_t)255;
        return p;
    };
    _Float16* x16    = (_Float16*)alloc((size_t)NN * 32 * 2);
    _Float16* ea16   = (_Float16*)alloc((size_t)EE * 32 * 2);        // padded K 4->32
    _Float16* wPre1  = (_Float16*)alloc((size_t)128 * 32 * 2);
    _Float16* wPre2  = (_Float16*)alloc((size_t)128 * 128 * 2);
    _Float16* wEnc1  = (_Float16*)alloc((size_t)256 * 32 * 2);       // padded K 4->32
    _Float16* wEnc2  = (_Float16*)alloc((size_t)256 * 256 * 2);
    _Float16* wEnc3  = (_Float16*)alloc((size_t)128 * 256 * 2);
    _Float16* wEnc4  = (_Float16*)alloc((size_t)16384 * 128 * 2);
    _Float16* wGih   = (_Float16*)alloc((size_t)384 * 128 * 2);
    _Float16* wGhh   = (_Float16*)alloc((size_t)384 * 128 * 2);
    _Float16* wPrd1  = (_Float16*)alloc((size_t)1024 * 512 * 2);
    _Float16* wPrd2  = (_Float16*)alloc((size_t)512 * 1024 * 2);
    _Float16* wPrd3  = (_Float16*)alloc((size_t)16 * 512 * 2);       // padded N 8->16
    _Float16* act116 = (_Float16*)alloc((size_t)NN * 128 * 2);
    float*    h32    = (float*)   alloc((size_t)NN * 128 * 4);
    _Float16* h16    = (_Float16*)alloc((size_t)NN * 128 * 2);
    _Float16* e116   = (_Float16*)alloc((size_t)EE * 256 * 2);
    _Float16* e216   = (_Float16*)alloc((size_t)EE * 256 * 2);
    _Float16* e316   = (_Float16*)alloc((size_t)EE * 128 * 2);
    _Float16* wbig   = (_Float16*)alloc((size_t)EE * 16384 * 2);     // 256 MiB
    float*    deg    = (float*)   alloc((size_t)NN * 4);
    float*    agg    = (float*)   alloc((size_t)NN * 128 * 4);
    _Float16* m16    = (_Float16*)alloc((size_t)NN * 128 * 2);
    float*    gx     = (float*)   alloc((size_t)NN * 384 * 4);
    float*    gh     = (float*)   alloc((size_t)NN * 384 * 4);
    float*    hq     = (float*)   alloc((size_t)GG * 128 * 4);
    float*    cst    = (float*)   alloc((size_t)GG * 128 * 4);
    float*    qstar  = (float*)   alloc((size_t)GG * 256 * 4);
    float*    gates  = (float*)   alloc((size_t)GG * 512 * 4);
    float*    evec   = (float*)   alloc((size_t)NN * 4);
    int*      gstart = (int*)     alloc((size_t)GG * 4);
    int*      gend   = (int*)     alloc((size_t)GG * 4);
    _Float16* feat16 = (_Float16*)alloc((size_t)NN * 512 * 2);
    _Float16* p116   = (_Float16*)alloc((size_t)NN * 1024 * 2);
    _Float16* p216   = (_Float16*)alloc((size_t)NN * 512 * 2);
    float*    p3     = (float*)   alloc((size_t)NN * 16 * 4);

    // ---- stage f16 copies (zero-padded) ----
    k_cvt_f16<<<g1(NN * 32), 256, 0, stream>>>(x, x16, NN, 32, NN, 32);
    k_cvt_f16<<<g1(EE * 32), 256, 0, stream>>>(edge_attr, ea16, EE, 32, EE, 4);
    k_cvt_f16<<<g1(128 * 32), 256, 0, stream>>>(pre_W1, wPre1, 128, 32, 128, 32);
    k_cvt_f16<<<g1(128 * 128), 256, 0, stream>>>(pre_W2, wPre2, 128, 128, 128, 128);
    k_cvt_f16<<<g1(256 * 32), 256, 0, stream>>>(enc_W1, wEnc1, 256, 32, 256, 4);
    k_cvt_f16<<<g1(256 * 256), 256, 0, stream>>>(enc_W2, wEnc2, 256, 256, 256, 256);
    k_cvt_f16<<<g1(128 * 256), 256, 0, stream>>>(enc_W3, wEnc3, 128, 256, 128, 256);
    k_cvt_f16<<<g1(16384 * 128), 256, 0, stream>>>(enc_W4, wEnc4, 16384, 128, 16384, 128);
    k_cvt_f16<<<g1(384 * 128), 256, 0, stream>>>(gru_Wih, wGih, 384, 128, 384, 128);
    k_cvt_f16<<<g1(384 * 128), 256, 0, stream>>>(gru_Whh, wGhh, 384, 128, 384, 128);
    k_cvt_f16<<<g1(1024 * 512), 256, 0, stream>>>(pred_W1, wPrd1, 1024, 512, 1024, 512);
    k_cvt_f16<<<g1(512 * 1024), 256, 0, stream>>>(pred_W2, wPrd2, 512, 1024, 512, 1024);
    k_cvt_f16<<<g1(16 * 512), 256, 0, stream>>>(pred_W3, wPrd3, 16, 512, 8, 512);

    // ---- preprocess MLP: x -> h ----
    launch_gemm(stream, x16, wPre1, nullptr, act116, pre1g, pre1b, pre1m, pre1v, nullptr,
                NN, 128, 32, 1);
    launch_gemm(stream, act116, wPre2, h32, h16, pre2g, pre2b, pre2m, pre2v, nullptr,
                NN, 128, 128, 1);

    // ---- edge network -> per-edge weights (f16) ----
    launch_gemm(stream, ea16, wEnc1, nullptr, e116, enc1g, enc1b, enc1m, enc1v, nullptr,
                EE, 256, 32, 1);
    launch_gemm(stream, e116, wEnc2, nullptr, e216, enc2g, enc2b, enc2m, enc2v, nullptr,
                EE, 256, 256, 1);
    launch_gemm(stream, e216, wEnc3, nullptr, e316, enc3g, enc3b, enc3m, enc3v, nullptr,
                EE, 128, 256, 1);
    launch_gemm(stream, e316, wEnc4, nullptr, wbig, enc4g, enc4b, enc4m, enc4v, nullptr,
                EE, 16384, 128, 0);

    // ---- degree ----
    k_zero_f32<<<g1(NN), 256, 0, stream>>>(deg, NN);
    k_deg<<<g1(EE), 256, 0, stream>>>(e_dst, deg, EE);

    // ---- 3x message passing + GRU ----
    for (int it = 0; it < 3; ++it) {
        k_zero_f32<<<g1(NN * 128), 256, 0, stream>>>(agg, NN * 128);
        k_msg_scatter<<<EE / 8, 256, 0, stream>>>(h32, wbig, e_src, e_dst, agg, EE);
        k_agg_finalize<<<g1(NN * 128), 256, 0, stream>>>(agg, deg, conv_bias, m16, NN * 128);
        launch_gemm(stream, m16, wGih, gx, nullptr, nullptr, nullptr, nullptr, nullptr,
                    gru_bih, NN, 384, 128, 0);
        launch_gemm(stream, h16, wGhh, gh, nullptr, nullptr, nullptr, nullptr, nullptr,
                    gru_bhh, NN, 384, 128, 0);
        k_gru_combine<<<g1(NN * 128), 256, 0, stream>>>(gx, gh, h32, h16, NN * 128);
    }

    // ---- Set2Set pooling ----
    k_zero_f32<<<g1(GG * 128), 256, 0, stream>>>(hq, GG * 128);
    k_zero_f32<<<g1(GG * 128), 256, 0, stream>>>(cst, GG * 128);
    k_zero_f32<<<g1(GG * 256), 256, 0, stream>>>(qstar, GG * 256);
    k_bounds_init<<<g1(GG), 256, 0, stream>>>(gstart, gend, GG);
    k_bounds_find<<<g1(NN), 256, 0, stream>>>(batch, gstart, gend, NN);
    for (int it = 0; it < 3; ++it) {
        k_lstm_gates<<<g1(GG * 512), 256, 0, stream>>>(qstar, hq, lstm_Wih, lstm_Whh,
                                                       lstm_bih, lstm_bhh, gates);
        k_lstm_update<<<g1(GG * 128), 256, 0, stream>>>(gates, cst, hq, GG * 128);
        k_attn_e<<<NN / 8, 256, 0, stream>>>(h32, hq, batch, evec, NN);
        k_set2set_reduce<<<GG, 128, 0, stream>>>(evec, h32, hq, gstart, gend, qstar);
    }

    // ---- prediction head ----
    k_feat_gather<<<g1(NN * 512), 256, 0, stream>>>(h32, qstar, t0, t1, batch, feat16, NN * 512);
    launch_gemm(stream, feat16, wPrd1, nullptr, p116, prd1g, prd1b, prd1m, prd1v, nullptr,
                NN, 1024, 512, 1);
    launch_gemm(stream, p116, wPrd2, nullptr, p216, prd2g, prd2b, prd2m, prd2v, nullptr,
                NN, 512, 1024, 1);
    launch_gemm(stream, p216, wPrd3, p3, nullptr, nullptr, nullptr, nullptr, nullptr,
                nullptr, NN, 16, 512, 0);
    k_final_gather<<<g1(NN), 256, 0, stream>>>(p3, pred_b3, tclass, (float*)d_out, NN);
}
